// MoEGate_74457553043955
// MI455X (gfx1250) — matrix operand order repaired and checked by reading, not compiled, not run on hardware
//
#include <hip/hip_runtime.h>
#include <hip/hip_bf16.h>
#include <cstdint>

typedef __attribute__((ext_vector_type(16))) _Float16 v16h;
typedef __attribute__((ext_vector_type(8)))  _Float16 v8h;
typedef __attribute__((ext_vector_type(8)))  float    v8f;

#define HID   7168
#define NEXP  256
#define TOKENS 8192
#define TOPK  8
#define TOPKGROUP 4
#define SCALE 2.5f

// ---------------------------------------------------------------------------
// Prep: transpose + convert gate weights fp32 [HID][NEXP] -> f16 BT[NEXP][HID]
// Coalesced fp32 reads; each thread writes one 64B contiguous f16 chunk.
// ---------------------------------------------------------------------------
__global__ __launch_bounds__(256) void bt_convert(const float* __restrict__ Bk,
                                                  _Float16* __restrict__ BT)
{
    const int n  = threadIdx.x;        // expert column 0..255
    const int k0 = blockIdx.x * 32;    // 224 blocks cover K=7168
    _Float16 tmp[32];
#pragma unroll
    for (int i = 0; i < 32; ++i)
        tmp[i] = (_Float16)Bk[(size_t)(k0 + i) * NEXP + n];
    _Float16* dst = BT + (size_t)n * HID + k0;
#pragma unroll
    for (int c = 0; c < 4; ++c) {
        v8h v;
#pragma unroll
        for (int j = 0; j < 8; ++j) v[j] = tmp[c * 8 + j];
        *(v8h*)(dst + c * 8) = v;
    }
}

// ---------------------------------------------------------------------------
// GEMM: logits[8192][256] = A[8192][7168] (fp32, cvt->f16) x BT^T, f32 acc.
// Wave tile: M=32 x N=256, K step 32 via v_wmma_f32_16x16x32_f16.
// A-fragment (16-bit A 16x32): lanes 0-15 rows M=0..15, elems0-7=K0..7,
// elems8-15=K16..23; lanes 16-31 same rows, K+8. B-fragment (32x16): lane&15
// = column, lanes<16 hold K0..7,16..23, lanes>=16 hold K8..15,24..31 (same order as A).
// ---------------------------------------------------------------------------
__device__ __forceinline__ v16h load_a_frag(const float* __restrict__ p)
{
    const float4* q = (const float4*)p;
    float4 x0 = q[0], x1 = q[1];   // K +0..7   (relative to lane base)
    float4 y0 = q[4], y1 = q[5];   // K +16..23
    v16h a;
    a[0]  = (_Float16)x0.x; a[1]  = (_Float16)x0.y;
    a[2]  = (_Float16)x0.z; a[3]  = (_Float16)x0.w;
    a[4]  = (_Float16)x1.x; a[5]  = (_Float16)x1.y;
    a[6]  = (_Float16)x1.z; a[7]  = (_Float16)x1.w;
    a[8]  = (_Float16)y0.x; a[9]  = (_Float16)y0.y;
    a[10] = (_Float16)y0.z; a[11] = (_Float16)y0.w;
    a[12] = (_Float16)y1.x; a[13] = (_Float16)y1.y;
    a[14] = (_Float16)y1.z; a[15] = (_Float16)y1.w;
    return a;
}

__global__ __launch_bounds__(64) void gate_gemm(const float* __restrict__ A,
                                                const _Float16* __restrict__ BT,
                                                float* __restrict__ logits)
{
    const int lane = threadIdx.x & 31;
    const int wave = threadIdx.x >> 5;
    const int col  = lane & 15;
    const int half = lane >> 4;

    const int rowBase = blockIdx.x * 64 + wave * 32;
    const float* a0p = A + (size_t)(rowBase + col) * HID + half * 8;
    const float* a1p = a0p + (size_t)16 * HID;
    const _Float16* btp = BT + (size_t)col * HID + half * 8;

    v8f acc0[16], acc1[16];
#pragma unroll
    for (int i = 0; i < 16; ++i) { acc0[i] = v8f{}; acc1[i] = v8f{}; }

    for (int kb = 0; kb < HID; kb += 32) {
        v16h a0 = load_a_frag(a0p + kb);
        v16h a1 = load_a_frag(a1p + kb);
        const _Float16* bk = btp + kb;
#pragma unroll
        for (int nt = 0; nt < 16; ++nt) {
            const _Float16* bp = bk + (size_t)nt * 16 * HID;
            v8h b0 = *(const v8h*)bp;
            v8h b1 = *(const v8h*)(bp + 16);
            v16h b = __builtin_shufflevector(b0, b1,
                     0,1,2,3,4,5,6,7,8,9,10,11,12,13,14,15);
            acc0[nt] = __builtin_amdgcn_wmma_f32_16x16x32_f16(
                false, a0, false, b, (short)0, acc0[nt], false, false);
            acc1[nt] = __builtin_amdgcn_wmma_f32_16x16x32_f16(
                false, a1, false, b, (short)0, acc1[nt], false, false);
        }
    }

    // C layout: VGPR r -> M=r (lanes 0-15) / M=r+8 (lanes 16-31), N=lane&15
    const int rOut = rowBase + half * 8;
#pragma unroll
    for (int nt = 0; nt < 16; ++nt) {
        const int c = nt * 16 + col;
#pragma unroll
        for (int r = 0; r < 8; ++r) {
            logits[(size_t)(rOut + r) * NEXP + c]      = acc0[nt][r];
            logits[(size_t)(rOut + 16 + r) * NEXP + c] = acc1[nt][r];
        }
    }
}

// ---------------------------------------------------------------------------
// Routing: one wave32 per token. 8 experts/lane, group = lane>>2 (32 experts).
// ---------------------------------------------------------------------------
__global__ __launch_bounds__(256) void gate_route(const float* __restrict__ logits,
                                                  const float* __restrict__ bias,
                                                  float* __restrict__ out,
                                                  int tokens)
{
    const int lane  = threadIdx.x & 31;
    const int token = blockIdx.x * 8 + (threadIdx.x >> 5);
    if (token >= tokens) return;

    const float* row = logits + (size_t)token * NEXP;
    float4 c0 = ((const float4*)row)[lane * 2];
    float4 c1 = ((const float4*)row)[lane * 2 + 1];
    float4 b0 = ((const float4*)bias)[lane * 2];
    float4 b1 = ((const float4*)bias)[lane * 2 + 1];

    float s[8];
    s[0] = 1.f / (1.f + __expf(-c0.x)) + b0.x;
    s[1] = 1.f / (1.f + __expf(-c0.y)) + b0.y;
    s[2] = 1.f / (1.f + __expf(-c0.z)) + b0.z;
    s[3] = 1.f / (1.f + __expf(-c0.w)) + b0.w;
    s[4] = 1.f / (1.f + __expf(-c1.x)) + b1.x;
    s[5] = 1.f / (1.f + __expf(-c1.y)) + b1.y;
    s[6] = 1.f / (1.f + __expf(-c1.z)) + b1.z;
    s[7] = 1.f / (1.f + __expf(-c1.w)) + b1.w;

    // local top-2, then merge across the 4 lanes of the group
    float a1v = -1e30f, a2v = -1e30f;
#pragma unroll
    for (int j = 0; j < 8; ++j) {
        float v = s[j];
        if (v > a1v) { a2v = a1v; a1v = v; } else if (v > a2v) a2v = v;
    }
#pragma unroll
    for (int off = 1; off < 4; off <<= 1) {
        float o1 = __shfl_xor(a1v, off, 32);
        float o2 = __shfl_xor(a2v, off, 32);
        float m1 = fmaxf(a1v, o1);
        float m2 = fmaxf(fminf(a1v, o1), fmaxf(a2v, o2));
        a1v = m1; a2v = m2;
    }
    float gsum = a1v + a2v;
    const int myg = lane >> 2;
    int rank = 0;
#pragma unroll
    for (int g = 0; g < 8; ++g) {
        float og = __shfl(gsum, g * 4, 32);
        rank += (og > gsum) || (og == gsum && g < myg);
    }
    if (rank >= TOPKGROUP) {
#pragma unroll
        for (int j = 0; j < 8; ++j) s[j] = 0.0f;   // reference masks to 0.0
    }

    // 8 rounds of wave-wide argmax (lowest-index tiebreak = jax top_k order)
    float keep = 0.f, sum = 0.f;
#pragma unroll
    for (int it = 0; it < TOPK; ++it) {
        float lv = -1e30f; int li = 0;
#pragma unroll
        for (int j = 0; j < 8; ++j)
            if (s[j] > lv) { lv = s[j]; li = j; }
        int gi = lane * 8 + li;
#pragma unroll
        for (int off = 16; off >= 1; off >>= 1) {
            float ov = __shfl_xor(lv, off, 32);
            int   oi = __shfl_xor(gi, off, 32);
            if (ov > lv || (ov == lv && oi < gi)) { lv = ov; gi = oi; }
        }
        sum += lv;
        if (lane == it) keep = lv;
        if ((gi >> 3) == lane) s[gi & 7] = -1e30f;  // remove winner
    }
    float w = keep * SCALE / (sum + 1e-20f);
    if (lane < TOPK) out[(size_t)token * TOPK + lane] = w;
}

// ---------------------------------------------------------------------------
extern "C" void kernel_launch(void* const* d_in, const int* in_sizes, int n_in,
                              void* d_out, int out_size, void* d_ws, size_t ws_size,
                              hipStream_t stream)
{
    const float* A    = (const float*)d_in[0];   // hidden_states [8192,7168] f32
    const float* Bk   = (const float*)d_in[1];   // kernel        [7168, 256] f32
    const float* bias = (const float*)d_in[2];   // bias          [256]       f32
    float* out = (float*)d_out;                  // [8192, 8] f32

    _Float16* BT  = (_Float16*)d_ws;                         // 3.67 MB
    float* logits = (float*)((char*)d_ws + (4u << 20));      // 8 MB

    bt_convert<<<HID / 32, 256, 0, stream>>>(Bk, BT);
    gate_gemm<<<TOKENS / 64, 64, 0, stream>>>(A, BT, logits);
    gate_route<<<TOKENS / 8, 256, 0, stream>>>(logits, bias, out, TOKENS);
}
